// AFAA_31860067401960
// MI455X (gfx1250) — matrix operand order repaired and checked by reading, not compiled, not run on hardware
//
#include <hip/hip_runtime.h>
#include <hip/hip_bf16.h>

// ---------------------------------------------------------------------------
// Fused WDT-attention pipeline for MI455X (gfx1250, wave32, WMMA).
//   K1: depthwise 3x3 conv + Haar DWT  -> yL/yH (f32 outputs) + xcat (bf16, ws)
//   K2: qkv GEMM (192x256xN) via v_wmma_f32_16x16x32_bf16, LDS-resident weights,
//       fused BN+ReLU; writes v (bf16, ws) and per-block Gram/norm partials.
//   K3: reduce partials, softmax, fold attn+residual+proj -> Weff[b] (64x64 bf16)
//   K4: out = Weff @ v + proj_b via WMMA bf16.
// ---------------------------------------------------------------------------

typedef __bf16 bf16_t;
typedef bf16_t v16bf __attribute__((ext_vector_type(16)));
typedef float  v8f   __attribute__((ext_vector_type(8)));

union FragU { v16bf v; uint4 q[2]; };

static __device__ __forceinline__ unsigned short f2bf(float f) {
  unsigned int u = __float_as_uint(f);
  u += 0x7FFFu + ((u >> 16) & 1u);          // round-to-nearest-even
  return (unsigned short)(u >> 16);
}

// ---- geometry ----
#define CH      64
#define CO      192            // 3*CH
#define KDIM    256            // 4*CH
#define HH      256
#define NPB     65536          // HH*HH per batch
#define NTOT    131072         // 2 batches

// ---- d_out layout (floats) ----
#define OUT_OFF 0u
#define YL_OFF  8388608u
#define YH_OFF  16777216u

// ---- workspace layout (bytes) ----
#define XCAT_OFF  0ull                 // u16 [2][256][65536]   67,108,864 B
#define VBUF_OFF  67108864ull          // u16 [2][64][65536]    16,777,216 B
#define GPART_OFF 83886080ull          // f32 [2048][8][80]      5,242,880 B
#define WEFF_OFF  89128960ull          // u16 [2][64][64]           16,384 B

// ===========================================================================
// K1: depthwise conv 3x3 (SAME) + bias + Haar DWT.
// One thread per (b,c,oh,ow) 2x2 output quad.  8,388,608 threads.
// ===========================================================================
__global__ __launch_bounds__(256) void k1_dwt(
    const float* __restrict__ x, const float* __restrict__ dww,
    const float* __restrict__ dwb, float* __restrict__ out,
    unsigned short* __restrict__ xcat)
{
  int idx = blockIdx.x * 256 + threadIdx.x;
  int ow = idx & 255;
  int oh = (idx >> 8) & 255;
  int c  = (idx >> 16) & 63;
  int b  = idx >> 22;

  float wgt[9];
#pragma unroll
  for (int i = 0; i < 9; ++i) wgt[i] = dww[c * 9 + i];
  float bias = dwb[c];

  const float* xp = x + ((size_t)(b * CH + c)) * 512 * 512;
  int iy0 = oh * 2 - 1, ix0 = ow * 2 - 1;
  float xin[4][4];
#pragma unroll
  for (int dy = 0; dy < 4; ++dy) {
    int iy = iy0 + dy;
    bool yv = (unsigned)iy < 512u;
#pragma unroll
    for (int dx = 0; dx < 4; ++dx) {
      int ix = ix0 + dx;
      xin[dy][dx] = (yv && (unsigned)ix < 512u) ? xp[iy * 512 + ix] : 0.0f;
    }
  }
  float cv[2][2];
#pragma unroll
  for (int py = 0; py < 2; ++py)
#pragma unroll
    for (int px = 0; px < 2; ++px) {
      float s = bias;
#pragma unroll
      for (int i = 0; i < 3; ++i)
#pragma unroll
        for (int j = 0; j < 3; ++j)
          s = fmaf(wgt[i * 3 + j], xin[py + i][px + j], s);
      cv[py][px] = s;
    }
  float a = cv[0][0], bb = cv[0][1], cc = cv[1][0], dd = cv[1][1];
  float ll = (a + bb + cc + dd) * 0.5f;
  float lh = (a + bb - cc - dd) * 0.5f;
  float hl = (a - bb + cc - dd) * 0.5f;
  float hh = (a - bb - cc + dd) * 0.5f;

  size_t sp = (size_t)oh * 256 + ow;
  size_t cbase = ((size_t)(b * CH + c)) * NPB;
  out[YL_OFF + cbase + sp] = ll;
  size_t yh = YH_OFF + ((size_t)(b * CH + c) * 3) * NPB + sp;
  out[yh] = lh;
  out[yh + NPB] = hl;
  out[yh + 2 * NPB] = hh;

  size_t xb = ((size_t)b * KDIM + c) * NPB + sp;   // channel order: ll,lh,hl,hh
  xcat[xb] = f2bf(ll);
  xcat[xb + (size_t)CH * NPB * 1] = f2bf(lh);
  xcat[xb + (size_t)CH * NPB * 2] = f2bf(hl);
  xcat[xb + (size_t)CH * NPB * 3] = f2bf(hh);
}

// ===========================================================================
// K2: qkv = W(192x256) @ xcat + bias, BN affine, ReLU.  WMMA bf16.
// Block = 256 threads (8 waves), 64 N-columns per block (2048 blocks).
// Dynamic LDS: W bf16 [192][256] (96K) + Xt bf16 [64][264] (33K) + qk f32 (32K)
// Each wave: 6 output tiles (12 mt x 4 nt over 8 waves), 8 K-slices each
//   => 48 v_wmma_f32_16x16x32_bf16 per wave.
// q,k rows (0..127) go to LDS for the Gram partial; v rows go to ws (bf16).
// ===========================================================================
#define XST 264   // Xt row stride in u16 (16B aligned, padded)

__global__ __launch_bounds__(256) void k2_qkv(
    const unsigned short* __restrict__ xcat, const float* __restrict__ qkv_w,
    const float* __restrict__ qkv_b, const float* __restrict__ bn_g,
    const float* __restrict__ bn_bt, const float* __restrict__ bn_m,
    const float* __restrict__ bn_v, unsigned short* __restrict__ vbuf,
    float* __restrict__ gpart)
{
  extern __shared__ unsigned char smem[];
  unsigned short* Wl = (unsigned short*)smem;                 // 192*256
  unsigned short* Xt = Wl + CO * KDIM;                        // 64*XST
  float* qk = (float*)(smem + (CO * KDIM + 64 * XST) * 2);    // 128*64 f32

  const int t = threadIdx.x;
  const int lane = t & 31, w = t >> 5;
  const int bx = blockIdx.x;
  const int b = bx >> 10;
  const int nb0 = (bx & 1023) * 64;

  // ---- weights -> LDS (f32 -> bf16) ----
  for (int i = t; i < CO * KDIM; i += 256) Wl[i] = f2bf(qkv_w[i]);

  // ---- xcat slab -> LDS, transposed to [col][k] ----
  {
    const uint4* src = (const uint4*)(xcat + ((size_t)(b * KDIM + t)) * NPB + nb0);
    union { uint4 q[8]; unsigned short h[64]; } u;
#pragma unroll
    for (int j = 0; j < 8; ++j) u.q[j] = src[j];
#pragma unroll
    for (int n = 0; n < 64; ++n) Xt[n * XST + t] = u.h[n];
  }
  __syncthreads();

  const int m = lane & 15, hi = lane >> 4;
  for (int j = 0; j < 6; ++j) {
    int tile = w * 6 + j;
    int mt = tile >> 2, nt = tile & 3;
    v8f acc = {0.f, 0.f, 0.f, 0.f, 0.f, 0.f, 0.f, 0.f};
#pragma unroll
    for (int ks = 0; ks < 8; ++ks) {
      FragU af, bf;
      // A: row (mt*16+m), K in {hk..hk+7} U {hk+16..hk+23}, hk = 8*hi
      const uint4* ap = (const uint4*)(Wl + (mt * 16 + m) * KDIM + ks * 32 + hi * 8);
      af.q[0] = ap[0];
      af.q[1] = ap[2];
      // B: col (nt*16+m), K in {hk..hk+7} U {hk+16..hk+23}, hk = 8*hi
      const uint4* bp = (const uint4*)(Xt + (nt * 16 + m) * XST + ks * 32 + hi * 8);
      bf.q[0] = bp[0];
      bf.q[1] = bp[2];
      acc = __builtin_amdgcn_wmma_f32_16x16x32_bf16(
          false, af.v, false, bf.v, (short)0, acc, false, false);
    }
    // epilogue: +bias, BN affine, ReLU; scatter q/k to LDS, v to ws
    const int n = lane & 15;
#pragma unroll
    for (int r = 0; r < 8; ++r) {
      int o = mt * 16 + r + hi * 8;
      float sc = bn_g[o] * rsqrtf(bn_v[o] + 1e-5f);
      float sh = fmaf(qkv_b[o] - bn_m[o], sc, bn_bt[o]);
      float val = fmaxf(fmaf(acc[r], sc, sh), 0.0f);
      int colloc = nt * 16 + n;
      if (o < 128) {
        qk[o * 64 + colloc] = val;                                   // q,k
      } else {
        vbuf[((size_t)(b * CH + (o - 128))) * NPB + nb0 + colloc] = f2bf(val);
      }
    }
  }
  __syncthreads();

  // ---- per-head Gram/norm partials: wave w == head w ----
  {
    const int c0 = lane >> 3, d = lane & 7;
    float g0 = 0.f, g1 = 0.f, s = 0.f;
    const float* qr0 = qk + (w * 8 + c0) * 64;
    const float* qr1 = qk + (w * 8 + c0 + 4) * 64;
    const float* kr  = qk + (64 + w * 8 + d) * 64;
    const float* nr  = (lane < 8) ? (qk + (w * 8 + lane) * 64)
                                  : (qk + (64 + w * 8 + (lane - 8)) * 64);
    for (int col = 0; col < 64; ++col) {
      float kv = kr[col];
      g0 = fmaf(qr0[col], kv, g0);
      g1 = fmaf(qr1[col], kv, g1);
      if (lane < 16) { float nv = nr[col]; s = fmaf(nv, nv, s); }
    }
    float* p = gpart + ((size_t)bx * 8 + w) * 80;
    p[c0 * 8 + d] = g0;
    p[(c0 + 4) * 8 + d] = g1;
    if (lane < 8) p[64 + lane] = s;
    else if (lane < 16) p[72 + (lane - 8)] = s;
  }
}

// ===========================================================================
// K3: reduce partials -> cosine attn -> softmax -> Weff = proj_w @ (A+I).
// Single block, 128 threads (thread = (b, head, c) row, then (b, o) row).
// ===========================================================================
__global__ __launch_bounds__(128) void k3_attn(
    const float* __restrict__ gpart, const float* __restrict__ temp,
    const float* __restrict__ proj_w, unsigned short* __restrict__ weff)
{
  __shared__ float attnL[128 * 8];
  const int t = threadIdx.x;
  const int b = t >> 6, rem = t & 63, head = rem >> 3, c = rem & 7;

  float g[8] = {0, 0, 0, 0, 0, 0, 0, 0};
  float sk[8] = {0, 0, 0, 0, 0, 0, 0, 0};
  float sq = 0.f;
  for (int blk = 0; blk < 1024; ++blk) {
    const float* p = gpart + ((size_t)((b * 1024 + blk) * 8 + head)) * 80;
#pragma unroll
    for (int d = 0; d < 8; ++d) {
      g[d] += p[c * 8 + d];
      sk[d] += p[72 + d];
    }
    sq += p[64 + c];
  }
  float nq = fmaxf(sqrtf(sq), 1e-12f);
  float tp = temp[head];
  float a[8], mx = -1e30f;
#pragma unroll
  for (int d = 0; d < 8; ++d) {
    a[d] = g[d] / (nq * fmaxf(sqrtf(sk[d]), 1e-12f)) * tp;
    mx = fmaxf(mx, a[d]);
  }
  float s = 0.f;
#pragma unroll
  for (int d = 0; d < 8; ++d) { a[d] = __expf(a[d] - mx); s += a[d]; }
  float inv = 1.0f / s;
#pragma unroll
  for (int d = 0; d < 8; ++d) attnL[t * 8 + d] = a[d] * inv;
  __syncthreads();

  // Weff[b][o][cin] = proj_w[o][cin] + sum_cc proj_w[o][h*8+cc]*attn[h][cc][cin%8]
  const int o = rem;
  for (int cin = 0; cin < 64; ++cin) {
    int hin = cin >> 3, dl = cin & 7;
    float s2 = proj_w[o * 64 + cin];
    const float* al = attnL + (b * 64 + hin * 8) * 8 + dl;
#pragma unroll
    for (int cc = 0; cc < 8; ++cc)
      s2 = fmaf(proj_w[o * 64 + hin * 8 + cc], al[cc * 8], s2);
    weff[(b * 64 + o) * 64 + cin] = f2bf(s2);
  }
}

// ===========================================================================
// K4: out = Weff[b](64x64) @ v[b](64x65536) + proj_b.  WMMA bf16.
// Block = 256 threads (8 waves), 128 columns per block (1024 blocks).
// Each wave: 4 M-tiles x 1 N-tile x 2 K-slices => 8 wmma.
// ===========================================================================
#define VST 80   // Vt row stride in u16 (16B aligned, padded)

__global__ __launch_bounds__(256) void k4_proj(
    const unsigned short* __restrict__ weff, const unsigned short* __restrict__ vbuf,
    const float* __restrict__ proj_b, float* __restrict__ out)
{
  __shared__ unsigned short Wl[64 * 64];
  __shared__ unsigned short Vt[128 * VST];

  const int t = threadIdx.x, lane = t & 31, w = t >> 5;
  const int bx = blockIdx.x;
  const int b = bx >> 9;
  const int nb0 = (bx & 511) * 128;

  for (int i = t; i < 64 * 64; i += 256) Wl[i] = weff[b * 64 * 64 + i];
  {
    int r = t >> 2, part = t & 3;
    const uint4* src =
        (const uint4*)(vbuf + ((size_t)(b * CH + r)) * NPB + nb0 + part * 32);
    union { uint4 q[4]; unsigned short h[32]; } u;
#pragma unroll
    for (int j = 0; j < 4; ++j) u.q[j] = src[j];
#pragma unroll
    for (int j = 0; j < 32; ++j) Vt[(part * 32 + j) * VST + r] = u.h[j];
  }
  __syncthreads();

  const int m = lane & 15, hi = lane >> 4;
  for (int mt = 0; mt < 4; ++mt) {
    v8f acc = {0.f, 0.f, 0.f, 0.f, 0.f, 0.f, 0.f, 0.f};
#pragma unroll
    for (int ks = 0; ks < 2; ++ks) {
      FragU af, bf;
      const uint4* ap = (const uint4*)(Wl + (mt * 16 + m) * 64 + ks * 32 + hi * 8);
      af.q[0] = ap[0];
      af.q[1] = ap[2];
      const uint4* bp = (const uint4*)(Vt + (w * 16 + m) * VST + ks * 32 + hi * 8);
      bf.q[0] = bp[0];
      bf.q[1] = bp[2];
      acc = __builtin_amdgcn_wmma_f32_16x16x32_bf16(
          false, af.v, false, bf.v, (short)0, acc, false, false);
    }
    const int n = lane & 15;
#pragma unroll
    for (int r = 0; r < 8; ++r) {
      int o = mt * 16 + r + hi * 8;
      size_t col = (size_t)nb0 + w * 16 + n;
      out[OUT_OFF + ((size_t)(b * CH + o)) * NPB + col] = acc[r] + proj_b[o];
    }
  }
}

// ===========================================================================
extern "C" void kernel_launch(void* const* d_in, const int* in_sizes, int n_in,
                              void* d_out, int out_size, void* d_ws, size_t ws_size,
                              hipStream_t stream) {
  (void)in_sizes; (void)n_in; (void)out_size; (void)ws_size;
  const float* x      = (const float*)d_in[0];
  const float* dw_w   = (const float*)d_in[1];
  const float* dw_b   = (const float*)d_in[2];
  const float* qkv_w  = (const float*)d_in[3];
  const float* qkv_b  = (const float*)d_in[4];
  const float* bn_g   = (const float*)d_in[5];
  const float* bn_bt  = (const float*)d_in[6];
  const float* bn_m   = (const float*)d_in[7];
  const float* bn_v   = (const float*)d_in[8];
  const float* temp   = (const float*)d_in[9];
  const float* proj_w = (const float*)d_in[10];
  const float* proj_b = (const float*)d_in[11];
  float* out = (float*)d_out;

  unsigned char* ws = (unsigned char*)d_ws;
  unsigned short* xcat = (unsigned short*)(ws + XCAT_OFF);
  unsigned short* vbuf = (unsigned short*)(ws + VBUF_OFF);
  float*          gpar = (float*)(ws + GPART_OFF);
  unsigned short* weff = (unsigned short*)(ws + WEFF_OFF);

  k1_dwt<<<32768, 256, 0, stream>>>(x, dw_w, dw_b, out, xcat);

  // dyn LDS: 192*256*2 + 64*XST*2 + 128*64*4 = 98304 + 33792 + 32768
  const size_t k2_smem = (size_t)(CO * KDIM + 64 * XST) * 2 + 128 * 64 * 4;
  k2_qkv<<<2048, 256, k2_smem, stream>>>(xcat, qkv_w, qkv_b, bn_g, bn_bt,
                                         bn_m, bn_v, vbuf, gpar);

  k3_attn<<<1, 128, 0, stream>>>(gpar, temp, proj_w, weff);

  k4_proj<<<1024, 256, 0, stream>>>(weff, vbuf, proj_b, out);
}
